// Mambaformer_13151189860537
// MI455X (gfx1250) — matrix operand order repaired and checked by reading, not compiled, not run on hardware
//
#include <hip/hip_runtime.h>
#include <hip/hip_bf16.h>

// ---------------------------------------------------------------------------
// Mambaformer block for MI455X (gfx1250, wave32, WMMA).
// L=31744 tokens, C=64, DI=128, DS=16. All GEMMs via v_wmma_f32_16x16x32_bf16,
// register-blocked 4 n-tiles per wave, 4 waves (m-tiles) per workgroup.
// Selective scan done as a 3-stage chunked associative scan (62 chunks x 512).
// ---------------------------------------------------------------------------

#define KLB __launch_bounds__

constexpr int L      = 31744;   // 31*32*32
constexpr int CHUNK  = 512;
constexpr int NCH    = L / CHUNK;   // 62
constexpr int MT     = L / 16;      // 1984 M-tiles

typedef __bf16 v16bf __attribute__((ext_vector_type(16)));
typedef float  v8f   __attribute__((ext_vector_type(8)));

union AFrag { unsigned int u[8]; v16bf v; };
union CFrag { float f[8]; v8f v; };

__device__ __forceinline__ unsigned short f2bf(float f) {
    unsigned int u = __float_as_uint(f);
    unsigned int r = u + 0x7FFFu + ((u >> 16) & 1u);   // round-to-nearest-even
    return (unsigned short)(r >> 16);
}
__device__ __forceinline__ float sigm(float x) { return 1.f / (1.f + __expf(-x)); }

// ---------------------------------------------------------------------------
// Pack a row-major f32 weight [KDIM, NDIM] into WMMA B-fragment order (bf16):
// per (ktile, ntile, lane): 8 contiguous dwords; dword v = {K, K+1} with
// K=kt*32+(v<4 ? 2v+8*hi : 16+2(v-4)+8*hi) at N=nt*16+lh (matches A fragment order).
// ---------------------------------------------------------------------------
__global__ void KLB(32) pack_weight(const float* __restrict__ W,
                                    unsigned int* __restrict__ Bp,
                                    int NDIM, int NT)
{
    const int lane = threadIdx.x;
    const int hi = lane >> 4, lh = lane & 15;
    const int nt = blockIdx.x;
    const int kt = blockIdx.y;
#pragma unroll
    for (int v = 0; v < 8; ++v) {
        int k = kt * 32 + ((v < 4) ? (2 * v + 8 * hi) : (16 + 2 * (v - 4) + 8 * hi));
        int n = nt * 16 + lh;
        unsigned int lo = f2bf(W[(size_t)k * NDIM + n]);
        unsigned int hw = f2bf(W[(size_t)(k + 1) * NDIM + n]);
        Bp[(((size_t)kt * NT + nt) * 32 + lane) * 8 + v] = lo | (hw << 16);
    }
}

// ---------------------------------------------------------------------------
// WMMA GEMM: D[M,N] = A[M,K](bf16,row-major) * Bpacked + epilogue.
// 4 waves per block (adjacent m-tiles); each wave register-blocks NB n-tiles,
// so one A fragment feeds NB consecutive v_wmma ops per ktile.
// A-fragment per the 16-bit A layout:
//   lane<16: row m0+lh, K pairs {2v,2v+1} (v<4) and {16+2(v-4),...} (v>=4)
//   lane>=16: same rows, K offset +8 within each half.
// MODE 0: plain f32 store (ldout).  MODE 1: += resid[col*L+row] (transposed x).
// MODE 3: out[col*L+row] = acc + resid[row*64+col] (final transposed store).
// ---------------------------------------------------------------------------
template <int MODE, int KDIM, int NDIM, int NB>
__global__ void KLB(128) wmma_gemm(const unsigned short* __restrict__ A,
                                   const unsigned int* __restrict__ Bp,
                                   float* __restrict__ out, int ldout,
                                   const float* __restrict__ resid)
{
    constexpr int KT = KDIM / 32;
    constexpr int NT = NDIM / 16;
    const int lane = threadIdx.x & 31;
    const int wave = threadIdx.x >> 5;
    const int hi = lane >> 4, lh = lane & 15;
    const int m0 = (blockIdx.x * 4 + wave) * 16;
    const int ntb = blockIdx.y * NB;

    CFrag acc[NB];
#pragma unroll
    for (int j = 0; j < NB; ++j)
#pragma unroll
        for (int r = 0; r < 8; ++r) acc[j].f[r] = 0.f;

#pragma unroll
    for (int kt = 0; kt < KT; ++kt) {
        AFrag a;
        const unsigned short* arow = A + (size_t)(m0 + lh) * KDIM + kt * 32;
#pragma unroll
        for (int v = 0; v < 8; ++v) {
            int k = (v < 4) ? (2 * v + 8 * hi) : (16 + 2 * (v - 4) + 8 * hi);
            a.u[v] = *(const unsigned int*)(arow + k);
        }
#pragma unroll
        for (int j = 0; j < NB; ++j) {
            AFrag b;
            const unsigned int* bp =
                Bp + (((size_t)kt * NT + (ntb + j)) * 32 + lane) * 8;
#pragma unroll
            for (int v = 0; v < 8; ++v) b.u[v] = bp[v];
            acc[j].v = __builtin_amdgcn_wmma_f32_16x16x32_bf16(
                false, a.v, false, b.v, (short)0, acc[j].v, false, false);
        }
    }

#pragma unroll
    for (int j = 0; j < NB; ++j) {
        const int col = (ntb + j) * 16 + lh;
#pragma unroll
        for (int r = 0; r < 8; ++r) {
            const int row = m0 + r + 8 * hi;
            float val = acc[j].f[r];
            if (MODE == 0) {
                out[(size_t)row * ldout + col] = val;
            } else if (MODE == 1) {
                out[(size_t)row * ldout + col] = val + resid[(size_t)col * L + row];
            } else {  // MODE 3: transposed final store with residual
                out[(size_t)col * L + row] = val + resid[(size_t)row * 64 + col];
            }
        }
    }
}

// ---------------------------------------------------------------------------
// LayerNorm over C=64; one wave per token, 2 channels per lane.
// transposed=1: X[c*L + t] (raw input layout); 0: X[t*64 + c]. Output bf16.
// ---------------------------------------------------------------------------
__global__ void KLB(32) ln_kernel(const float* __restrict__ X,
                                  const float* __restrict__ w,
                                  const float* __restrict__ b,
                                  unsigned short* __restrict__ out,
                                  int transposed)
{
    const int t = blockIdx.x;
    const int lane = threadIdx.x;
    float v0, v1;
    if (transposed) {
        v0 = X[(size_t)lane * L + t];
        v1 = X[(size_t)(lane + 32) * L + t];
    } else {
        v0 = X[(size_t)t * 64 + lane];
        v1 = X[(size_t)t * 64 + lane + 32];
    }
    float s = v0 + v1;
#pragma unroll
    for (int o = 16; o; o >>= 1) s += __shfl_xor(s, o, 32);
    float mu = s * (1.f / 64.f);
    float d0 = v0 - mu, d1 = v1 - mu;
    float q = d0 * d0 + d1 * d1;
#pragma unroll
    for (int o = 16; o; o >>= 1) q += __shfl_xor(q, o, 32);
    float rstd = rsqrtf(q * (1.f / 64.f) + 1e-5f);
    out[(size_t)t * 64 + lane]      = f2bf(d0 * rstd * w[lane] + b[lane]);
    out[(size_t)t * 64 + lane + 32] = f2bf(d1 * rstd * w[lane + 32] + b[lane + 32]);
}

// ---------------------------------------------------------------------------
// Causal depthwise conv (4 taps) + ReLU + x-proj (dt/B/C) + delta softplus.
// One thread per DI channel, 4 tokens per block.
// xiz: [L,256] f32 (xi = cols 0..127, z = cols 128..255 from W_in GEMM).
// ---------------------------------------------------------------------------
__global__ void KLB(128) conv_xproj(const float* __restrict__ xiz,
                                    const float* __restrict__ conv_w,
                                    const float* __restrict__ W_xproj,
                                    const float* __restrict__ W_dt,
                                    const float* __restrict__ dt_bias,
                                    float* __restrict__ xc,
                                    float* __restrict__ delta,
                                    float* __restrict__ BC)
{
    __shared__ float s_xc[128];
    __shared__ float s_dt[4];
    const int d = threadIdx.x;
    const int tbase = blockIdx.x * 4;
    const float w0 = conv_w[d * 4 + 0], w1 = conv_w[d * 4 + 1];
    const float w2 = conv_w[d * 4 + 2], w3 = conv_w[d * 4 + 3];
    const float bias = dt_bias[d];

    for (int i = 0; i < 4; ++i) {
        const int t = tbase + i;
        float acc = w3 * xiz[(size_t)t * 256 + d];
        if (t >= 1) acc += w2 * xiz[(size_t)(t - 1) * 256 + d];
        if (t >= 2) acc += w1 * xiz[(size_t)(t - 2) * 256 + d];
        if (t >= 3) acc += w0 * xiz[(size_t)(t - 3) * 256 + d];
        float xcv = fmaxf(acc, 0.f);
        xc[(size_t)t * 128 + d] = xcv;
        s_xc[d] = xcv;
        __syncthreads();
        if (d < 36) {  // dbl = xc_row @ W_xproj  (cols 0..3 dt, 4..19 B, 20..35 C)
            float dot = 0.f;
            for (int j = 0; j < 128; ++j) dot = fmaf(s_xc[j], W_xproj[j * 36 + d], dot);
            if (d < 4) s_dt[d] = dot;
            else       BC[(size_t)t * 32 + (d - 4)] = dot;
        }
        __syncthreads();
        float dv = bias;
#pragma unroll
        for (int r = 0; r < 4; ++r) dv = fmaf(s_dt[r], W_dt[r * 128 + d], dv);
        delta[(size_t)t * 128 + d] = (dv > 20.f) ? dv : log1pf(__expf(dv));
        __syncthreads();
    }
}

// ---------------------------------------------------------------------------
// Selective scan, stage A: per-chunk aggregates (prod dA, acc of dBu).
// One block per chunk, 1024 threads, each thread 2 (d,s) pairs (d and d+64).
// ---------------------------------------------------------------------------
__global__ void KLB(1024) scan_stageA(const float* __restrict__ delta,
                                      const float* __restrict__ xc,
                                      const float* __restrict__ BC,
                                      const float* __restrict__ A_log,
                                      float* __restrict__ prodA,
                                      float* __restrict__ accB)
{
    const int c = blockIdx.x;
    const int tid = threadIdx.x;
    const int s = tid & 15;
    const int d0 = tid >> 4, d1 = d0 + 64;
    const float A0 = -__expf(A_log[d0 * 16 + s]);
    const float A1 = -__expf(A_log[d1 * 16 + s]);
    float p0 = 1.f, p1 = 1.f, b0 = 0.f, b1 = 0.f;
    const int t0 = c * CHUNK;
    for (int i = 0; i < CHUNK; ++i) {
        const int t = t0 + i;
        int tp = t + 16; if (tp >= L) tp = L - 1;
        __builtin_prefetch(&delta[(size_t)tp * 128 + d0], 0, 0);  // global_prefetch_b8
        float bv  = BC[(size_t)t * 32 + s];
        float dl0 = delta[(size_t)t * 128 + d0];
        float u0  = xc[(size_t)t * 128 + d0];
        float dl1 = delta[(size_t)t * 128 + d1];
        float u1  = xc[(size_t)t * 128 + d1];
        float a0 = __expf(dl0 * A0), a1 = __expf(dl1 * A1);
        b0 = fmaf(b0, a0, dl0 * u0 * bv);
        b1 = fmaf(b1, a1, dl1 * u1 * bv);
        p0 *= a0; p1 *= a1;
    }
    prodA[(size_t)c * 2048 + d0 * 16 + s] = p0;
    prodA[(size_t)c * 2048 + d1 * 16 + s] = p1;
    accB [(size_t)c * 2048 + d0 * 16 + s] = b0;
    accB [(size_t)c * 2048 + d1 * 16 + s] = b1;
}

// Stage B: sequential prefix over the 62 chunk aggregates (trivial work).
__global__ void KLB(1024) scan_stageB(const float* __restrict__ prodA,
                                      const float* __restrict__ accB,
                                      float* __restrict__ hinit)
{
    const int tid = threadIdx.x;
#pragma unroll
    for (int pi = 0; pi < 2; ++pi) {
        const int p = tid + pi * 1024;
        float h = 0.f;
        for (int c = 0; c < NCH; ++c) {
            hinit[(size_t)c * 2048 + p] = h;
            h = fmaf(h, prodA[(size_t)c * 2048 + p], accB[(size_t)c * 2048 + p]);
        }
    }
}

// Stage C: replay each chunk with injected initial state, reduce over s,
// fuse the SiLU(z) gate, emit bf16 y*silu(z) for the W_out GEMM.
__global__ void KLB(1024) scan_stageC(const float* __restrict__ delta,
                                      const float* __restrict__ xc,
                                      const float* __restrict__ BC,
                                      const float* __restrict__ A_log,
                                      const float* __restrict__ D_ssm,
                                      const float* __restrict__ xiz,
                                      const float* __restrict__ hinit,
                                      unsigned short* __restrict__ yz)
{
    const int c = blockIdx.x;
    const int tid = threadIdx.x;
    const int s = tid & 15;
    const int d0 = tid >> 4, d1 = d0 + 64;
    const float A0 = -__expf(A_log[d0 * 16 + s]);
    const float A1 = -__expf(A_log[d1 * 16 + s]);
    const float D0 = D_ssm[d0], D1 = D_ssm[d1];
    float h0 = hinit[(size_t)c * 2048 + d0 * 16 + s];
    float h1 = hinit[(size_t)c * 2048 + d1 * 16 + s];
    const int t0 = c * CHUNK;
    for (int i = 0; i < CHUNK; ++i) {
        const int t = t0 + i;
        float bv  = BC[(size_t)t * 32 + s];
        float cv  = BC[(size_t)t * 32 + 16 + s];
        float dl0 = delta[(size_t)t * 128 + d0];
        float u0  = xc[(size_t)t * 128 + d0];
        float dl1 = delta[(size_t)t * 128 + d1];
        float u1  = xc[(size_t)t * 128 + d1];
        h0 = fmaf(h0, __expf(dl0 * A0), dl0 * u0 * bv);
        h1 = fmaf(h1, __expf(dl1 * A1), dl1 * u1 * bv);
        float y0 = h0 * cv, y1 = h1 * cv;
#pragma unroll
        for (int o = 8; o; o >>= 1) {       // reduce over the 16 state lanes
            y0 += __shfl_xor(y0, o, 16);
            y1 += __shfl_xor(y1, o, 16);
        }
        if (s == 0) {
            float z0 = xiz[(size_t)t * 256 + 128 + d0];
            float z1 = xiz[(size_t)t * 256 + 128 + d1];
            yz[(size_t)t * 128 + d0] = f2bf((y0 + u0 * D0) * (z0 * sigm(z0)));
            yz[(size_t)t * 128 + d1] = f2bf((y1 + u1 * D1) * (z1 * sigm(z1)));
        }
    }
}

// FFN gate: hgate = silu(g) * v, g = buf[:, :128], v = buf[:, 128:].
__global__ void gate_kernel(const float* __restrict__ buf,
                            unsigned short* __restrict__ hg)
{
    const int idx = blockIdx.x * blockDim.x + threadIdx.x;
    if (idx >= L * 128) return;
    const int t = idx >> 7, d = idx & 127;
    float g = buf[(size_t)t * 256 + d];
    float v = buf[(size_t)t * 256 + 128 + d];
    hg[idx] = f2bf(v * (g * sigm(g)));
}

// ---------------------------------------------------------------------------
extern "C" void kernel_launch(void* const* d_in, const int* in_sizes, int n_in,
                              void* d_out, int out_size, void* d_ws, size_t ws_size,
                              hipStream_t stream)
{
    (void)in_sizes; (void)n_in; (void)out_size; (void)ws_size;
    const float* x       = (const float*)d_in[0];
    const float* ln1_w   = (const float*)d_in[1];
    const float* ln1_b   = (const float*)d_in[2];
    const float* W_in    = (const float*)d_in[3];
    const float* conv_w  = (const float*)d_in[4];
    const float* W_xproj = (const float*)d_in[5];
    const float* W_dt    = (const float*)d_in[6];
    const float* dt_bias = (const float*)d_in[7];
    const float* A_log   = (const float*)d_in[8];
    const float* D_ssm   = (const float*)d_in[9];
    const float* W_out   = (const float*)d_in[10];
    const float* ln2_w   = (const float*)d_in[11];
    const float* ln2_b   = (const float*)d_in[12];
    const float* W_ffn1  = (const float*)d_in[13];
    const float* W_ffn2  = (const float*)d_in[14];
    float* out = (float*)d_out;

    char* w = (char*)d_ws;
    size_t off = 0;
    auto alloc = [&](size_t bytes) -> void* {
        void* p = w + off;
        off = (off + bytes + 255) & ~(size_t)255;
        return p;
    };
    unsigned short* xn1   = (unsigned short*)alloc((size_t)L * 64 * 2);
    float*          xiz   = (float*)alloc((size_t)L * 256 * 4);   // reused for FFN1 out
    float*          xc    = (float*)alloc((size_t)L * 128 * 4);
    float*          delta = (float*)alloc((size_t)L * 128 * 4);
    float*          BC    = (float*)alloc((size_t)L * 32 * 4);
    float*          prodA = (float*)alloc((size_t)NCH * 2048 * 4);
    float*          accB  = (float*)alloc((size_t)NCH * 2048 * 4);
    float*          hinit = (float*)alloc((size_t)NCH * 2048 * 4);
    unsigned short* yz    = (unsigned short*)alloc((size_t)L * 128 * 2);
    float*          xres  = (float*)alloc((size_t)L * 64 * 4);
    unsigned short* xn2   = (unsigned short*)alloc((size_t)L * 64 * 2);
    unsigned short* hg    = (unsigned short*)alloc((size_t)L * 128 * 2);
    unsigned int*   pin   = (unsigned int*)alloc(8192 * 4);
    unsigned int*   pout  = (unsigned int*)alloc(4096 * 4);
    unsigned int*   pff1  = (unsigned int*)alloc(8192 * 4);
    unsigned int*   pff2  = (unsigned int*)alloc(4096 * 4);

    // --- weight packing (tiny; reused by all 1984 M-tiles) ---
    pack_weight<<<dim3(16, 2), 32, 0, stream>>>(W_in,   pin, 256, 16);
    pack_weight<<<dim3(4, 4),  32, 0, stream>>>(W_out,  pout, 64, 4);
    pack_weight<<<dim3(16, 2), 32, 0, stream>>>(W_ffn1, pff1, 256, 16);
    pack_weight<<<dim3(4, 4),  32, 0, stream>>>(W_ffn2, pff2, 64, 4);

    // --- mixer branch ---
    ln_kernel<<<L, 32, 0, stream>>>(x, ln1_w, ln1_b, xn1, 1);
    wmma_gemm<0, 64, 256, 4><<<dim3(MT / 4, 4), 128, 0, stream>>>(xn1, pin, xiz, 256, nullptr);
    conv_xproj<<<L / 4, 128, 0, stream>>>(xiz, conv_w, W_xproj, W_dt, dt_bias,
                                          xc, delta, BC);
    scan_stageA<<<NCH, 1024, 0, stream>>>(delta, xc, BC, A_log, prodA, accB);
    scan_stageB<<<1, 1024, 0, stream>>>(prodA, accB, hinit);
    scan_stageC<<<NCH, 1024, 0, stream>>>(delta, xc, BC, A_log, D_ssm, xiz, hinit, yz);
    wmma_gemm<1, 128, 64, 4><<<dim3(MT / 4, 1), 128, 0, stream>>>(yz, pout, xres, 64, x);

    // --- gated FFN branch ---
    ln_kernel<<<L, 32, 0, stream>>>(xres, ln2_w, ln2_b, xn2, 0);
    wmma_gemm<0, 64, 256, 4><<<dim3(MT / 4, 4), 128, 0, stream>>>(xn2, pff1, xiz, 256, nullptr);
    gate_kernel<<<(L * 128 + 255) / 256, 256, 0, stream>>>(xiz, hg);
    wmma_gemm<3, 128, 64, 4><<<dim3(MT / 4, 1), 128, 0, stream>>>(hg, pff2, out, 64, xres);
}
